// MultiRelationalBlock_69363721831020
// MI455X (gfx1250) — matrix operand order repaired and checked by reading, not compiled, not run on hardware
//
#include <hip/hip_runtime.h>

// ---------------------------------------------------------------------------
// MultiRelationalBlock for MI455X (gfx1250, wave32, WMMA bf16 f32-accumulate)
// Round 2: + async global->LDS staging of GEMM A-strips (ASYNCcnt path),
//          + global_prefetch of relation-bias strips in fused attention.
// ---------------------------------------------------------------------------

typedef __bf16 bf16;
typedef __attribute__((ext_vector_type(16))) __bf16 v16bf;
typedef __attribute__((ext_vector_type(8)))  __bf16 v8bf;
typedef __attribute__((ext_vector_type(8)))  float  v8f;

#define BB      8
#define NTOK    1024
#define DMODEL  256
#define NHEAD   8
#define DHEAD   32
#define DFFN    1024
#define MTOT    (BB * NTOK)          // 8192 token rows

__device__ __forceinline__ bf16 to_bf16(float f) {
  unsigned u = __builtin_bit_cast(unsigned, f);
  u += 0x7FFFu + ((u >> 16) & 1u);            // round-to-nearest-even
  unsigned short h = (unsigned short)(u >> 16);
  return __builtin_bit_cast(bf16, h);
}

__device__ __forceinline__ v16bf cat8(v8bf lo, v8bf hi) {
  return __builtin_shufflevector(lo, hi, 0, 1, 2, 3, 4, 5, 6, 7,
                                         8, 9, 10, 11, 12, 13, 14, 15);
}

// Async copy of one 16-byte chunk: global (saddr + 32-bit voffset) -> LDS.
// Tracked by ASYNCcnt; no VGPR data movement.
__device__ __forceinline__ void async_g2l_b128(unsigned lds_byte_off,
                                               const void* sbase,
                                               unsigned g_byte_off) {
  asm volatile("global_load_async_to_lds_b128 %0, %1, %2"
               :
               : "v"(lds_byte_off), "v"(g_byte_off), "s"(sbase)
               : "memory");
}

__device__ __forceinline__ void wait_async_all() {
  asm volatile("s_wait_asynccnt 0x0" ::: "memory");
}

// ---------------------------------------------------------------------------
// fp32 -> bf16 conversion
// ---------------------------------------------------------------------------
__global__ __launch_bounds__(256) void cvt_bf16_kernel(const float* __restrict__ src,
                                                       bf16* __restrict__ dst, int n) {
  int i = blockIdx.x * 256 + threadIdx.x;
  if (i < n) dst[i] = to_bf16(src[i]);
}

// ---------------------------------------------------------------------------
// QKV projection: C[8192,768] = X[8192,256] @ Wqkv[256,768] (+bias),
// scattered straight into Q[b][h][t][dh], Kt[b][h][dh][t], V[b][h][t][dh].
// One wave = one 16x16 tile; block = 8 waves = 16x128 strip.
// A-strip (16x256 bf16 = 8KB) async-staged to LDS, shared by all 8 waves.
// ---------------------------------------------------------------------------
__global__ __launch_bounds__(256)
void qkv_gemm_kernel(const bf16* __restrict__ A, const bf16* __restrict__ W,
                     const float* __restrict__ bq, const float* __restrict__ bk,
                     const float* __restrict__ bv,
                     bf16* __restrict__ Qb, bf16* __restrict__ Ktb, bf16* __restrict__ Vb) {
  __shared__ bf16 sA[16 * DMODEL];                         // 8 KB
  const int tid = threadIdx.x;
  const int wave = tid >> 5, lane = tid & 31;
  const int half = lane >> 4, mcol = lane & 15;
  const int m0 = blockIdx.x * 16;
  const int ncol = blockIdx.y * 128 + wave * 16;           // 0..767
  const int which = ncol >> 8, f0 = ncol & 255;            // 0=Q,1=K,2=V

  // async-stage A strip: 16 rows x 256 bf16 = 512 x 16B chunks
  {
    const bf16* Abase = A + (size_t)m0 * DMODEL;
    const unsigned sbase = (unsigned)(size_t)&sA[0];
    for (int i = tid; i < 2 * DMODEL; i += 256)
      async_g2l_b128(sbase + i * 16u, Abase, i * 16u);
    wait_async_all();
  }
  __syncthreads();

  const bf16* arow = sA + mcol * DMODEL;
  const bf16* Bm = W + (size_t)which * DMODEL * DMODEL + f0;
  v8f c = {};
  for (int k0 = 0; k0 < DMODEL; k0 += 32) {
    v16bf af = cat8(*(const v8bf*)(arow + k0 + half * 8),
                    *(const v8bf*)(arow + k0 + 16 + half * 8));
    const bf16* bcol = Bm + (size_t)k0 * DMODEL + mcol;
    v16bf bfr;
    for (int t = 0; t < 8; ++t) {
      bfr[t]     = bcol[(size_t)(half * 8 + t) * DMODEL];
      bfr[t + 8] = bcol[(size_t)(16 + half * 8 + t) * DMODEL];
    }
    c = __builtin_amdgcn_wmma_f32_16x16x32_bf16(false, af, false, bfr,
                                                (short)0, c, false, false);
  }
  const int f = f0 + mcol;
  const float bias = (which == 0) ? bq[f] : (which == 1) ? bk[f] : bv[f];
  const int hh = f >> 5, dh = f & 31;
#pragma unroll
  for (int r = 0; r < 8; ++r) {
    const int m = m0 + r + half * 8;
    const int bb = m >> 10, t = m & (NTOK - 1);
    const bf16 v = to_bf16(c[r] + bias);
    if (which == 0)
      Qb[(((size_t)(bb * NHEAD + hh)) * NTOK + t) * DHEAD + dh] = v;
    else if (which == 1)
      Ktb[(((size_t)(bb * NHEAD + hh)) * DHEAD + dh) * NTOK + t] = v;
    else
      Vb[(((size_t)(bb * NHEAD + hh)) * NTOK + t) * DHEAD + dh] = v;
  }
}

// ---------------------------------------------------------------------------
// Fused attention: scores (WMMA) -> +rel_bias/mask -> softmax -> attn write
// -> attn @ V (WMMA) -> Y (concat-head bf16).
// Block = (b, h, 16-row tile); 8 waves. LDS = exactly 64KB, triple-reused:
//   phase 1/2: scores fp32 [16][1024]; phase 3: probabilities fp32;
//   phase 4: per-wave partial Y [8][16*32].
// ---------------------------------------------------------------------------
__global__ __launch_bounds__(256)
void attn_fused_kernel(const bf16* __restrict__ Qb, const bf16* __restrict__ Ktb,
                       const bf16* __restrict__ Vb,
                       const float* __restrict__ adj, const float* __restrict__ attr,
                       const float* __restrict__ rep, const float* __restrict__ mask,
                       float* __restrict__ attn_out, bf16* __restrict__ Yb) {
  __shared__ float sS[16 * NTOK];                          // 64 KB
  const int tid = threadIdx.x;
  const int wave = tid >> 5, lane = tid & 31;
  const int half = lane >> 4, mcol = lane & 15;
  const int b = blockIdx.z, h = blockIdx.y;
  const int m0 = blockIdx.x * 16;

  const size_t headOff = ((size_t)(b * NHEAD + h)) * NTOK * DHEAD;
  const bf16* Qh = Qb + headOff;                           // [NTOK][32]
  const bf16* Vh = Vb + headOff;                           // [NTOK][32]
  const bf16* Kth = Ktb + ((size_t)(b * NHEAD + h)) * DHEAD * NTOK;  // [32][NTOK]

  // Prefetch the 3 x 64KB relation-bias strips so HBM reads overlap the
  // score WMMA phase (global_prefetch_b8, no counters, no VGPR return).
  const size_t rel0 = (size_t)b * NTOK * NTOK + (size_t)m0 * NTOK;
  for (int i = tid; i < 512; i += 256) {                   // 512 x 128B lines
    __builtin_prefetch((const char*)(adj + rel0) + i * 128, 0, 1);
    __builtin_prefetch((const char*)(attr + rel0) + i * 128, 0, 1);
    __builtin_prefetch((const char*)(rep + rel0) + i * 128, 0, 1);
  }

  // --- Q fragment (16 rows x K=32, loaded once) ---
  const bf16* qrow = Qh + (size_t)(m0 + mcol) * DHEAD;
  const v16bf afrag = cat8(*(const v8bf*)(qrow + half * 8),
                           *(const v8bf*)(qrow + 16 + half * 8));
  const float scale = 0.1767766952966369f;                 // 1/sqrt(32)

  // --- scores: each wave covers 8 key tiles (128 keys); 1 WMMA per tile ---
#pragma unroll
  for (int j = 0; j < 8; ++j) {
    const int n0 = (wave * 8 + j) * 16;
    const bf16* kcol = Kth + n0 + mcol;
    v16bf bfrag;
    for (int t = 0; t < 8; ++t) {
      bfrag[t]     = kcol[(size_t)(half * 8 + t) * NTOK];
      bfrag[t + 8] = kcol[(size_t)(16 + half * 8 + t) * NTOK];
    }
    v8f c = {};
    c = __builtin_amdgcn_wmma_f32_16x16x32_bf16(false, afrag, false, bfrag,
                                                (short)0, c, false, false);
#pragma unroll
    for (int r = 0; r < 8; ++r)
      sS[(r + half * 8) * NTOK + n0 + mcol] = c[r] * scale;
  }
  __syncthreads();

  // --- relation bias + key mask ---
  for (int i = tid; i < 16 * NTOK; i += 256) {
    const int n = i & (NTOK - 1);
    const size_t g = rel0 + (size_t)(i >> 10) * NTOK + n;
    float v = sS[i] + adj[g] + attr[g] + rep[g];
    sS[i] = (mask[(size_t)b * NTOK + n] > 0.0f) ? v : -1.0e9f;
  }
  __syncthreads();

  // --- softmax: 2 rows per wave; write attn (fp32, a block output) ---
#pragma unroll
  for (int rr = 0; rr < 2; ++rr) {
    const int r = wave * 2 + rr;
    float* row = sS + r * NTOK;
    float mx = -3.0e38f;
    for (int n = lane; n < NTOK; n += 32) mx = fmaxf(mx, row[n]);
#pragma unroll
    for (int off = 16; off > 0; off >>= 1) mx = fmaxf(mx, __shfl_xor(mx, off, 32));
    float sum = 0.0f;
    for (int n = lane; n < NTOK; n += 32) {
      float e = __expf(row[n] - mx);
      row[n] = e;
      sum += e;
    }
#pragma unroll
    for (int off = 16; off > 0; off >>= 1) sum += __shfl_xor(sum, off, 32);
    const float inv = 1.0f / sum;
    float* aout = attn_out + (((size_t)(b * NHEAD + h)) * NTOK + (m0 + r)) * NTOK;
    for (int n = lane; n < NTOK; n += 32) {
      const float p = row[n] * inv;
      row[n] = p;
      aout[n] = p;
    }
  }
  __syncthreads();

  // --- attn @ V: wave w owns key range [w*128, w*128+128), 4 K-steps,
  //     two 16x16 output tiles (dh 0..15 and 16..31) ---
  v8f c0 = {}, c1 = {};
#pragma unroll
  for (int ks = 0; ks < 4; ++ks) {
    const int kb = wave * 128 + ks * 32;
    const float* prow = sS + mcol * NTOK + kb;
    v16bf pf;
#pragma unroll
    for (int t = 0; t < 8; ++t) {
      pf[t]     = to_bf16(prow[half * 8 + t]);
      pf[t + 8] = to_bf16(prow[16 + half * 8 + t]);
    }
    const bf16* vrow = Vh + (size_t)kb * DHEAD + mcol;
    v16bf b0, b1;
    for (int t = 0; t < 8; ++t) {
      b0[t]     = vrow[(half * 8 + t) * DHEAD];
      b0[t + 8] = vrow[(16 + half * 8 + t) * DHEAD];
      b1[t]     = vrow[(half * 8 + t) * DHEAD + 16];
      b1[t + 8] = vrow[(16 + half * 8 + t) * DHEAD + 16];
    }
    c0 = __builtin_amdgcn_wmma_f32_16x16x32_bf16(false, pf, false, b0,
                                                 (short)0, c0, false, false);
    c1 = __builtin_amdgcn_wmma_f32_16x16x32_bf16(false, pf, false, b1,
                                                 (short)0, c1, false, false);
  }
  __syncthreads();                       // all probability reads complete
  float* ypart = sS;                     // reuse as [8][16*32] partials
#pragma unroll
  for (int r = 0; r < 8; ++r) {
    ypart[wave * 512 + (r + half * 8) * 32 + mcol]      = c0[r];
    ypart[wave * 512 + (r + half * 8) * 32 + 16 + mcol] = c1[r];
  }
  __syncthreads();
  for (int i = tid; i < 512; i += 256) {
    float s = 0.0f;
#pragma unroll
    for (int w = 0; w < 8; ++w) s += ypart[w * 512 + i];
    const int r = i >> 5, dh = i & 31;
    Yb[((size_t)b * NTOK + m0 + r) * DMODEL + h * DHEAD + dh] = to_bf16(s);
  }
}

// ---------------------------------------------------------------------------
// Generic bf16 GEMM + bias (+optional ReLU); fp32 and/or bf16 output.
// One wave = one 16x16 tile; block = 16x128 strip.
// A-strip (16xK bf16, <=32KB) async-staged to LDS, shared by all 8 waves.
// ---------------------------------------------------------------------------
template <int RELU, int WF, int WB>
__global__ __launch_bounds__(256)
void gemm_bias_kernel(const bf16* __restrict__ A, const bf16* __restrict__ Bm,
                      const float* __restrict__ bias, float* __restrict__ Cf,
                      bf16* __restrict__ Cb, int N, int K) {
  __shared__ bf16 sA[16 * 1024];                           // up to 32 KB used
  const int tid = threadIdx.x;
  const int wave = tid >> 5, lane = tid & 31;
  const int half = lane >> 4, mcol = lane & 15;
  const int m0 = blockIdx.x * 16;
  const int n0 = blockIdx.y * 128 + wave * 16;

  // async-stage A strip: 16 rows x K bf16 = 2K x 16B chunks
  {
    const bf16* Abase = A + (size_t)m0 * K;
    const unsigned sbase = (unsigned)(size_t)&sA[0];
    for (int i = tid; i < 2 * K; i += 256)
      async_g2l_b128(sbase + i * 16u, Abase, i * 16u);
    wait_async_all();
  }
  __syncthreads();

  const bf16* arow = sA + mcol * K;
  const bf16* bbase = Bm + n0;
  v8f c = {};
  for (int k0 = 0; k0 < K; k0 += 32) {
    v16bf af = cat8(*(const v8bf*)(arow + k0 + half * 8),
                    *(const v8bf*)(arow + k0 + 16 + half * 8));
    const bf16* bcol = bbase + (size_t)k0 * N + mcol;
    v16bf bfr;
    for (int t = 0; t < 8; ++t) {
      bfr[t]     = bcol[(size_t)(half * 8 + t) * N];
      bfr[t + 8] = bcol[(size_t)(16 + half * 8 + t) * N];
    }
    c = __builtin_amdgcn_wmma_f32_16x16x32_bf16(false, af, false, bfr,
                                                (short)0, c, false, false);
  }
  const float bv = bias[n0 + mcol];
#pragma unroll
  for (int r = 0; r < 8; ++r) {
    float v = c[r] + bv;
    if (RELU) v = fmaxf(v, 0.0f);
    const size_t idx = (size_t)(m0 + r + half * 8) * N + n0 + mcol;
    if (WF) Cf[idx] = v;
    if (WB) Cb[idx] = to_bf16(v);
  }
}

// ---------------------------------------------------------------------------
// AddNorm: out = LN(X + Y) * g + b  (row = 256 elems, 1 thread each)
// ---------------------------------------------------------------------------
__global__ __launch_bounds__(256)
void addnorm_kernel(const float* __restrict__ X, const float* __restrict__ Yv,
                    const float* __restrict__ g, const float* __restrict__ bt,
                    float* __restrict__ outf, bf16* __restrict__ outb) {
  __shared__ float rs[8], rs2[8];
  const int row = blockIdx.x, tid = threadIdx.x;
  const size_t idx = (size_t)row * DMODEL + tid;
  const float v = X[idx] + Yv[idx];
  float s = v, s2 = v * v;
#pragma unroll
  for (int off = 16; off > 0; off >>= 1) {
    s += __shfl_xor(s, off, 32);
    s2 += __shfl_xor(s2, off, 32);
  }
  if ((tid & 31) == 0) { rs[tid >> 5] = s; rs2[tid >> 5] = s2; }
  __syncthreads();
  float ts = 0.0f, ts2 = 0.0f;
#pragma unroll
  for (int w = 0; w < 8; ++w) { ts += rs[w]; ts2 += rs2[w]; }
  const float mu = ts * (1.0f / DMODEL);
  const float var = ts2 * (1.0f / DMODEL) - mu * mu;
  const float inv = rsqrtf(var + 1e-5f);
  const float o = (v - mu) * inv * g[tid] + bt[tid];
  outf[idx] = o;
  if (outb) outb[idx] = to_bf16(o);
}

// ---------------------------------------------------------------------------
// Workspace layout (bytes). HB aliases the dead Q/Kt/V/Y region; FFN2 output
// aliases YO (consumed by AddNorm1 before FFN2 runs). Total ~41.5 MiB.
// ---------------------------------------------------------------------------
static constexpr size_t SZ_XB = (size_t)MTOT * DMODEL * 2;               // 4 MiB
static constexpr size_t SZ_W = (size_t)DMODEL * DMODEL * 2;              // 128 KiB
static constexpr size_t SZ_WF = (size_t)DMODEL * DFFN * 2;               // 512 KiB
static constexpr size_t SZ_HEADS = (size_t)BB * NHEAD * NTOK * DHEAD * 2;// 4 MiB
static constexpr size_t OFF_XB = 0;
static constexpr size_t OFF_WQKV = OFF_XB + SZ_XB;
static constexpr size_t OFF_WO = OFF_WQKV + 3 * SZ_W;
static constexpr size_t OFF_WF1 = OFF_WO + SZ_W;
static constexpr size_t OFF_WF2 = OFF_WF1 + SZ_WF;
static constexpr size_t OFF_QB = OFF_WF2 + SZ_WF;
static constexpr size_t OFF_KT = OFF_QB + SZ_HEADS;
static constexpr size_t OFF_VB = OFF_KT + SZ_HEADS;
static constexpr size_t OFF_YB = OFF_VB + SZ_HEADS;
static constexpr size_t OFF_YO = OFF_YB + SZ_XB;                         // fp32 8 MiB
static constexpr size_t OFF_X1F = OFF_YO + (size_t)MTOT * DMODEL * 4;
static constexpr size_t OFF_X1B = OFF_X1F + (size_t)MTOT * DMODEL * 4;
static constexpr size_t OFF_HB = OFF_QB;                                 // alias, 16 MiB
static constexpr size_t OFF_FO = OFF_YO;                                 // alias

extern "C" void kernel_launch(void* const* d_in, const int* in_sizes, int n_in,
                              void* d_out, int out_size, void* d_ws, size_t ws_size,
                              hipStream_t stream) {
  (void)in_sizes; (void)n_in; (void)out_size; (void)ws_size;
  const float* X    = (const float*)d_in[0];
  const float* mask = (const float*)d_in[1];
  const float* adj  = (const float*)d_in[2];
  const float* attr = (const float*)d_in[3];
  const float* rep  = (const float*)d_in[4];
  const float* Wq = (const float*)d_in[5];
  const float* bq = (const float*)d_in[6];
  const float* Wk = (const float*)d_in[7];
  const float* bk = (const float*)d_in[8];
  const float* Wv = (const float*)d_in[9];
  const float* bv = (const float*)d_in[10];
  const float* Wo = (const float*)d_in[11];
  const float* bo = (const float*)d_in[12];
  const float* Wf1 = (const float*)d_in[13];
  const float* bf1 = (const float*)d_in[14];
  const float* Wf2 = (const float*)d_in[15];
  const float* bf2 = (const float*)d_in[16];
  const float* g1 = (const float*)d_in[17];
  const float* b1 = (const float*)d_in[18];
  const float* g2 = (const float*)d_in[19];
  const float* b2 = (const float*)d_in[20];

  char* ws = (char*)d_ws;
  bf16* XB   = (bf16*)(ws + OFF_XB);
  bf16* WQKV = (bf16*)(ws + OFF_WQKV);
  bf16* WOB  = (bf16*)(ws + OFF_WO);
  bf16* WF1B = (bf16*)(ws + OFF_WF1);
  bf16* WF2B = (bf16*)(ws + OFF_WF2);
  bf16* QB   = (bf16*)(ws + OFF_QB);
  bf16* KTB  = (bf16*)(ws + OFF_KT);
  bf16* VB   = (bf16*)(ws + OFF_VB);
  bf16* YB   = (bf16*)(ws + OFF_YB);
  float* YO  = (float*)(ws + OFF_YO);
  float* X1F = (float*)(ws + OFF_X1F);
  bf16* X1B  = (bf16*)(ws + OFF_X1B);
  bf16* HB   = (bf16*)(ws + OFF_HB);
  float* FO  = (float*)(ws + OFF_FO);

  float* X2out = (float*)d_out;
  float* attn_out = (float*)d_out + (size_t)MTOT * DMODEL;   // after X2

  // --- fp32 -> bf16 staging ---
  cvt_bf16_kernel<<<dim3((MTOT * DMODEL + 255) / 256), 256, 0, stream>>>(X, XB, MTOT * DMODEL);
  cvt_bf16_kernel<<<dim3(256), 256, 0, stream>>>(Wq, WQKV, DMODEL * DMODEL);
  cvt_bf16_kernel<<<dim3(256), 256, 0, stream>>>(Wk, WQKV + DMODEL * DMODEL, DMODEL * DMODEL);
  cvt_bf16_kernel<<<dim3(256), 256, 0, stream>>>(Wv, WQKV + 2 * DMODEL * DMODEL, DMODEL * DMODEL);
  cvt_bf16_kernel<<<dim3(256), 256, 0, stream>>>(Wo, WOB, DMODEL * DMODEL);
  cvt_bf16_kernel<<<dim3(1024), 256, 0, stream>>>(Wf1, WF1B, DMODEL * DFFN);
  cvt_bf16_kernel<<<dim3(1024), 256, 0, stream>>>(Wf2, WF2B, DFFN * DMODEL);

  // --- QKV projection ---
  qkv_gemm_kernel<<<dim3(MTOT / 16, 6), 256, 0, stream>>>(XB, WQKV, bq, bk, bv, QB, KTB, VB);

  // --- fused attention (scores+softmax+attn-write+AV) ---
  attn_fused_kernel<<<dim3(NTOK / 16, NHEAD, BB), 256, 0, stream>>>(
      QB, KTB, VB, adj, attr, rep, mask, attn_out, YB);

  // --- output projection ---
  gemm_bias_kernel<0, 1, 0><<<dim3(MTOT / 16, DMODEL / 128), 256, 0, stream>>>(
      YB, WOB, bo, YO, (bf16*)nullptr, DMODEL, DMODEL);

  // --- AddNorm 1 ---
  addnorm_kernel<<<dim3(MTOT), 256, 0, stream>>>(X, YO, g1, b1, X1F, X1B);

  // --- FFN ---
  gemm_bias_kernel<1, 0, 1><<<dim3(MTOT / 16, DFFN / 128), 256, 0, stream>>>(
      X1B, WF1B, bf1, (float*)nullptr, HB, DFFN, DMODEL);
  gemm_bias_kernel<0, 1, 0><<<dim3(MTOT / 16, DMODEL / 128), 256, 0, stream>>>(
      HB, WF2B, bf2, FO, (bf16*)nullptr, DMODEL, DFFN);

  // --- AddNorm 2 -> X2 ---
  addnorm_kernel<<<dim3(MTOT), 256, 0, stream>>>(X1F, FO, g2, b2, X2out, (bf16*)nullptr);
}
